// SpatialRelationEncoder_15822659518549
// MI455X (gfx1250) — matrix operand order repaired and checked by reading, not compiled, not run on hardware
//
#include <hip/hip_runtime.h>
#include <hip/hip_bf16.h>

// ---------------------------------------------------------------------------
// SpatialRelationEncoder for MI455X (gfx1250, wave32, WMMA, bf16/f32-accum).
//
// Exact linear refactor:
//   h[b,i,e] = sum_k yS[b,k,i,:] @ WkS[k,:,e]   (stacked [4608x8192]x[8192x1024])
//            + sum_k c[b,i,k]*qW[b,k,e] + s[b,i,k]*bk[k,e]
//   yS[b,k,i,:] = sum_j Akn[b,i,j,k] * spa_v[b,j,:]
//   qW[b,k,:]   = q[b,:] @ Wk[k,1024:,:]        (step-invariant)
//
// All B operands are stored TRANSPOSED (BT[n][k], k contiguous) so both A and
// B WMMA fragments load as b128 vectors: 2 loads per fragment, 2 loads/WMMA
// with 32x32-per-wave register blocking.
// ---------------------------------------------------------------------------

#define B_    128
#define N_    36
#define VDIM  2048
#define QDIM  1024
#define ODIM  1024
#define KG    8
#define EDIM  64
#define LDIM  11
#define INDIM 2048
#define BN    (B_ * N_)        // 4608
#define EPSF  1e-8f

typedef __attribute__((ext_vector_type(16))) __bf16 bf16x16;
typedef __attribute__((ext_vector_type(8)))  float  floatx8;

union FragU { bf16x16 h; unsigned int u[8]; };

// A fragment (16-bit, 16x32, MxK), ISA table layout:
//   lane l: row m = l&15; khi = (l>=16)?8:0
//   vgpr 0..3 -> k in [khi, khi+8), vgpr 4..7 -> k in [16+khi, 16+khi+8)
// rowp points at this lane's A row (k contiguous). Two b128 loads.
__device__ __forceinline__ bf16x16 load_a_rowptr(const __bf16* rowp) {
  FragU f;
  const int lane = threadIdx.x & 31;
  const int khi  = ((lane >> 4) & 1) << 3;
  const uint4 a = *reinterpret_cast<const uint4*>(rowp + khi);
  const uint4 b = *reinterpret_cast<const uint4*>(rowp + khi + 16);
  f.u[0] = a.x; f.u[1] = a.y; f.u[2] = a.z; f.u[3] = a.w;
  f.u[4] = b.x; f.u[5] = b.y; f.u[6] = b.z; f.u[7] = b.w;
  return f.h;
}

// B fragment (16-bit, 32x16, KxN) from TRANSPOSED storage BT[n][k]:
//   lane l: col n = l&15; kh = (l>=16)?8:0; vgpr 0..3 -> k in [kh, kh+8), vgpr 4..7 -> k in [16+kh, 16+kh+8)
// colp = BT + n*ld + kstep (caller adds lane column). Two b128 loads.
__device__ __forceinline__ bf16x16 load_bT_colptr(const __bf16* colp) {
  FragU f;
  const int lane = threadIdx.x & 31;
  const int kh   = ((lane >> 4) & 1) << 3;
  const uint4 a = *reinterpret_cast<const uint4*>(colp + kh);
  const uint4 b = *reinterpret_cast<const uint4*>(colp + kh + 16);
  f.u[0] = a.x; f.u[1] = a.y; f.u[2] = a.z; f.u[3] = a.w;
  f.u[4] = b.x; f.u[5] = b.y; f.u[6] = b.z; f.u[7] = b.w;
  return f.h;
}

__device__ __forceinline__ floatx8 wmma_bf16(bf16x16 a, bf16x16 b, floatx8 c) {
  return __builtin_amdgcn_wmma_f32_16x16x32_bf16(false, a, false, b, (short)0, c,
                                                 false, false);
}

// C tile coords: col n = n0 + (lane&15); row m = m0 + v + 8*(lane>=16)
__device__ __forceinline__ void epi_coords(int& nloc, int& mh) {
  const int lane = threadIdx.x & 31;
  nloc = lane & 15;
  mh   = ((lane >> 4) & 1) << 3;
}

// --------------------------------------------------------------------------
// GEMM 1: spa_v = relu(v_bf16 [4608x2048] @ Wv + b_v); BT = WvT [1024x2048].
// 32x32 per wave.  grid (144, 4) x 256
// --------------------------------------------------------------------------
__global__ void k_gemm_vtrans(const __bf16* __restrict__ vbf,
                              const __bf16* __restrict__ wvT,
                              const float*  __restrict__ bv,
                              float*        __restrict__ out) {
  const int wave = threadIdx.x >> 5;
  const int m0   = blockIdx.x * 32;
  const int n0   = (blockIdx.y * 8 + wave) * 32;
  int nloc, mh; epi_coords(nloc, mh);
  const int rl = nloc;
  floatx8 a00 = {}, a01 = {}, a10 = {}, a11 = {};
  const __bf16* ar0 = vbf + (size_t)(m0 + rl) * VDIM;
  const __bf16* ar1 = ar0 + (size_t)16 * VDIM;
  const __bf16* bc0 = wvT + (size_t)(n0 + rl) * VDIM;
  const __bf16* bc1 = bc0 + (size_t)16 * VDIM;
  for (int k0 = 0; k0 < VDIM; k0 += 32) {
    bf16x16 fa0 = load_a_rowptr(ar0 + k0);
    bf16x16 fa1 = load_a_rowptr(ar1 + k0);
    bf16x16 fb0 = load_bT_colptr(bc0 + k0);
    bf16x16 fb1 = load_bT_colptr(bc1 + k0);
    a00 = wmma_bf16(fa0, fb0, a00);
    a01 = wmma_bf16(fa0, fb1, a01);
    a10 = wmma_bf16(fa1, fb0, a10);
    a11 = wmma_bf16(fa1, fb1, a11);
  }
  const floatx8 accs[4] = {a00, a01, a10, a11};
#pragma unroll
  for (int tm = 0; tm < 2; ++tm)
#pragma unroll
    for (int tn = 0; tn < 2; ++tn) {
      const floatx8 acc = accs[tm * 2 + tn];
      const int n = n0 + tn * 16 + nloc;
      const float bias = bv[n];
#pragma unroll
      for (int v = 0; v < 8; ++v) {
        const int m = m0 + tm * 16 + v + mh;
        float val = acc[v] + bias;
        out[(size_t)m * ODIM + n] = val > 0.f ? val : 0.f;
      }
    }
}

// --------------------------------------------------------------------------
// GEMM 2: qW[b,g,:] = q_bf16 [128x1024] @ Wk[g,1024:,:]; BT slice of WkT.
// 32x32 per wave.  grid (4, 4, KG) x 256
// --------------------------------------------------------------------------
__global__ void k_gemm_qw(const __bf16* __restrict__ qbf,
                          const __bf16* __restrict__ wkT,   // [g][e:1024][d:2048]
                          float*        __restrict__ qw) {
  const int wave = threadIdx.x >> 5;
  const int m0   = blockIdx.x * 32;
  const int n0   = (blockIdx.y * 8 + wave) * 32;
  const int g    = blockIdx.z;
  int nloc, mh; epi_coords(nloc, mh);
  const int rl = nloc;
  floatx8 a00 = {}, a01 = {}, a10 = {}, a11 = {};
  const __bf16* ar0 = qbf + (size_t)(m0 + rl) * QDIM;
  const __bf16* ar1 = ar0 + (size_t)16 * QDIM;
  const __bf16* bc0 = wkT + ((size_t)g * 1024 + n0 + rl) * INDIM + 1024; // q half
  const __bf16* bc1 = bc0 + (size_t)16 * INDIM;
  for (int k0 = 0; k0 < QDIM; k0 += 32) {
    bf16x16 fa0 = load_a_rowptr(ar0 + k0);
    bf16x16 fa1 = load_a_rowptr(ar1 + k0);
    bf16x16 fb0 = load_bT_colptr(bc0 + k0);
    bf16x16 fb1 = load_bT_colptr(bc1 + k0);
    a00 = wmma_bf16(fa0, fb0, a00);
    a01 = wmma_bf16(fa0, fb1, a01);
    a10 = wmma_bf16(fa1, fb0, a10);
    a11 = wmma_bf16(fa1, fb1, a11);
  }
  const floatx8 accs[4] = {a00, a01, a10, a11};
#pragma unroll
  for (int tm = 0; tm < 2; ++tm)
#pragma unroll
    for (int tn = 0; tn < 2; ++tn) {
      const floatx8 acc = accs[tm * 2 + tn];
      const int n = n0 + tn * 16 + nloc;
#pragma unroll
      for (int v = 0; v < 8; ++v) {
        const int m = m0 + tm * 16 + v + mh;
        qw[((size_t)m * KG + g) * ODIM + n] = acc[v];
      }
    }
}

// --------------------------------------------------------------------------
// GEMM 3 (per step): yS[b,g,i,:] = Akn[b,g][48x64] @ spa[b][64x1024]
// BT = spaT[b][e:1024][j:64].  16x32 per wave. grid (3, 4, B_*KG) x 256
// --------------------------------------------------------------------------
__global__ void k_gemm_agg(const __bf16* __restrict__ akn,
                           const __bf16* __restrict__ spaT,
                           __bf16*       __restrict__ ys) {
  const int wave = threadIdx.x >> 5;
  const int m0   = blockIdx.x * 16;
  const int n0   = (blockIdx.y * 8 + wave) * 32;
  const int bg   = blockIdx.z;          // b*KG + g
  const int b    = bg >> 3;
  int nloc, mh; epi_coords(nloc, mh);
  const int rl = nloc;
  floatx8 a0 = {}, a1 = {};
  const __bf16* ar  = akn + ((size_t)bg * 48 + m0 + rl) * 64;
  const __bf16* bc0 = spaT + ((size_t)b * ODIM + n0 + rl) * 64;
  const __bf16* bc1 = bc0 + (size_t)16 * 64;
#pragma unroll
  for (int k0 = 0; k0 < 64; k0 += 32) {
    bf16x16 fa  = load_a_rowptr(ar + k0);
    bf16x16 fb0 = load_bT_colptr(bc0 + k0);
    bf16x16 fb1 = load_bT_colptr(bc1 + k0);
    a0 = wmma_bf16(fa, fb0, a0);
    a1 = wmma_bf16(fa, fb1, a1);
  }
  const floatx8 accs[2] = {a0, a1};
#pragma unroll
  for (int tn = 0; tn < 2; ++tn) {
    const floatx8 acc = accs[tn];
    const int n = n0 + tn * 16 + nloc;
#pragma unroll
    for (int v = 0; v < 8; ++v) {
      const int i = m0 + v + mh;
      if (i < N_) ys[((size_t)bg * N_ + i) * ODIM + n] = (__bf16)acc[v];
    }
  }
}

// --------------------------------------------------------------------------
// GEMM 4 (per step): h = yS_flat [4608 x 8192] @ WkS; spa += relu(h + hq)
// BT = WkT[g][e][d<1024].  32x32 per wave.  grid (144, 4) x 256
// --------------------------------------------------------------------------
__global__ void k_gemm_proj(const __bf16* __restrict__ ys,
                            const __bf16* __restrict__ wkT,
                            const float*  __restrict__ hq,
                            float*        __restrict__ spa) {
  const int wave = threadIdx.x >> 5;
  const int m0   = blockIdx.x * 32;
  const int n0   = (blockIdx.y * 8 + wave) * 32;
  int nloc, mh; epi_coords(nloc, mh);
  const int rl = nloc;
  const int r0 = m0 + rl, r1 = r0 + 16;
  const __bf16* ar0 = ys + ((size_t)(r0 / N_) * KG * N_ + (r0 % N_)) * ODIM;
  const __bf16* ar1 = ys + ((size_t)(r1 / N_) * KG * N_ + (r1 % N_)) * ODIM;
  floatx8 a00 = {}, a01 = {}, a10 = {}, a11 = {};
  for (int g = 0; g < KG; ++g) {
    const __bf16* arg0 = ar0 + (size_t)g * N_ * ODIM;
    const __bf16* arg1 = ar1 + (size_t)g * N_ * ODIM;
    const __bf16* bc0  = wkT + ((size_t)g * 1024 + n0 + rl) * INDIM; // spa half
    const __bf16* bc1  = bc0 + (size_t)16 * INDIM;
    for (int d0 = 0; d0 < ODIM; d0 += 32) {
      bf16x16 fa0 = load_a_rowptr(arg0 + d0);
      bf16x16 fa1 = load_a_rowptr(arg1 + d0);
      bf16x16 fb0 = load_bT_colptr(bc0 + d0);
      bf16x16 fb1 = load_bT_colptr(bc1 + d0);
      a00 = wmma_bf16(fa0, fb0, a00);
      a01 = wmma_bf16(fa0, fb1, a01);
      a10 = wmma_bf16(fa1, fb0, a10);
      a11 = wmma_bf16(fa1, fb1, a11);
    }
  }
  const floatx8 accs[4] = {a00, a01, a10, a11};
#pragma unroll
  for (int tm = 0; tm < 2; ++tm)
#pragma unroll
    for (int tn = 0; tn < 2; ++tn) {
      const floatx8 acc = accs[tm * 2 + tn];
      const int n = n0 + tn * 16 + nloc;
#pragma unroll
      for (int v = 0; v < 8; ++v) {
        const int r2 = m0 + tm * 16 + v + mh;
        float h   = acc[v] + hq[(size_t)r2 * ODIM + n];
        float rel = h > 0.f ? h : 0.f;
        spa[(size_t)r2 * ODIM + n] += rel;
      }
    }
}

// ----------------------- small helper kernels -----------------------------

__global__ void k_w(const float* __restrict__ E, const float* __restrict__ mu,
                    const float* __restrict__ logs, float* __restrict__ wg) {
  __shared__ float uw[LDIM * KG];
  __shared__ float rs[LDIM];
  const int t = threadIdx.x;
  if (t < LDIM * KG) {
    const int l = t >> 3, k = t & 7;
    float d2 = 0.f;
    for (int e = 0; e < EDIM; ++e) {
      float d = E[l * EDIM + e] - mu[k * EDIM + e];
      d2 += d * d;
    }
    uw[t] = expf(-0.5f * d2 / expf(2.f * logs[k]));
  }
  __syncthreads();
  if (t < LDIM) {
    float s = 0.f;
    for (int k = 0; k < KG; ++k) s += uw[t * KG + k];
    rs[t] = s + EPSF;
  }
  __syncthreads();
  if (t < LDIM * KG) wg[t] = uw[t] / rs[t >> 3];
}

// Akn[b,g,i,j] bf16 padded [48][64] + s[b,i,g]. 288 thr/blk, one per (b,i).
__global__ void k_akn(const float* __restrict__ adj, const float* __restrict__ wg,
                      __bf16* __restrict__ akn, float* __restrict__ sg) {
  __shared__ float sAk[N_ * KG];
  __shared__ float inv[KG];
  const int r = blockIdx.x;             // b*36 + i
  const int b = r / N_, i = r % N_;
  const int t = threadIdx.x;            // j*8 + g
  const int j = t >> 3, g = t & 7;
  float ak = 0.f;
  const float* arow = adj + ((size_t)r * N_ + j) * LDIM;
#pragma unroll
  for (int l = 0; l < LDIM; ++l) ak += arow[l] * wg[l * KG + g];
  sAk[t] = ak;
  __syncthreads();
  if (t < KG) {
    float s = 0.f;
    for (int jj = 0; jj < N_; ++jj) s += sAk[jj * KG + t];
    inv[t] = 1.f / (s + EPSF);
    sg[(size_t)r * KG + t] = s / (s + EPSF);
  }
  __syncthreads();
  akn[(((size_t)b * KG + g) * 48 + i) * 64 + j] = (__bf16)(ak * inv[g]);
}

// mask[b,i] = (rowsum(spa_v[b,i,:]) != 0)
__global__ void k_mask(const float* __restrict__ spa, float* __restrict__ mask) {
  __shared__ float red[256];
  const int r = blockIdx.x;
  const int t = threadIdx.x;
  const float* row = spa + (size_t)r * ODIM;
  float s = 0.f;
#pragma unroll
  for (int e = t; e < ODIM; e += 256) s += row[e];
  red[t] = s;
  __syncthreads();
  for (int off = 128; off > 0; off >>= 1) {
    if (t < off) red[t] += red[t + off];
    __syncthreads();
  }
  if (t == 0) mask[r] = (red[0] != 0.f) ? 1.f : 0.f;
}

// spaT[b][e][j] = bf16(spa[b][j][e]), j padded to 64 with zeros.
// grid (B_, 16) x 256, LDS-staged 64-column tiles.
__global__ void k_spaT(const float* __restrict__ spa, __bf16* __restrict__ spaT) {
  __shared__ __bf16 tile[64][40];       // [e'][j], padded
  const int b  = blockIdx.x;
  const int e0 = blockIdx.y * 64;
  const int t  = threadIdx.x;
  for (int idx = t; idx < N_ * 64; idx += 256) {
    const int ep = idx & 63, j = idx >> 6;
    tile[ep][j] = (__bf16)spa[((size_t)b * N_ + j) * ODIM + e0 + ep];
  }
  __syncthreads();
  for (int idx = t; idx < 64 * 64; idx += 256) {
    const int j = idx & 63, ep = idx >> 6;
    spaT[((size_t)b * ODIM + e0 + ep) * 64 + j] =
        (j < N_) ? tile[ep][j] : (__bf16)0.f;
  }
}

// c[b,i,g] = sum_j Akn[b,g,i,j] * mask[b,j].  288 thr/blk, one block per b.
__global__ void k_c(const __bf16* __restrict__ akn, const float* __restrict__ mask,
                    float* __restrict__ cg) {
  const int b = blockIdx.x;
  const int t = threadIdx.x;            // i*8 + g
  const int i = t >> 3, g = t & 7;
  const __bf16* arow = akn + (((size_t)b * KG + g) * 48 + i) * 64;
  const float*  mrow = mask + (size_t)b * N_;
  float c = 0.f;
#pragma unroll
  for (int j = 0; j < N_; ++j) c += (float)arow[j] * mrow[j];
  cg[((size_t)b * N_ + i) * KG + g] = c;
}

// hq[r,e] = sum_g c[r,g]*qW[b,g,e] + s[r,g]*bk[g,e]
__global__ void k_hq(const float* __restrict__ cg, const float* __restrict__ sg,
                     const float* __restrict__ qw, const float* __restrict__ bk,
                     float* __restrict__ hq) {
  const int r = blockIdx.x;
  const int b = r / N_;
  const int t = threadIdx.x;
  float cl[KG], sl[KG];
#pragma unroll
  for (int g = 0; g < KG; ++g) {
    cl[g] = cg[(size_t)r * KG + g];
    sl[g] = sg[(size_t)r * KG + g];
  }
#pragma unroll
  for (int e = t; e < ODIM; e += 256) {
    float acc = 0.f;
#pragma unroll
    for (int g = 0; g < KG; ++g)
      acc += cl[g] * qw[((size_t)b * KG + g) * ODIM + e] +
             sl[g] * bk[(size_t)g * ODIM + e];
    hq[(size_t)r * ODIM + e] = acc;
  }
}

__global__ void k_cvt(const float* __restrict__ s, __bf16* __restrict__ d, size_t n) {
  size_t i = (size_t)blockIdx.x * blockDim.x + threadIdx.x;
  const size_t stride = (size_t)gridDim.x * blockDim.x;
  for (; i < n; i += stride) d[i] = (__bf16)s[i];
}

// dst[g][c][r] = bf16(src[g][r][c]); R,C multiples of 32. grid (R/32,C/32,G)x256
__global__ void k_cvtT(const float* __restrict__ src, __bf16* __restrict__ dst,
                       int R, int C) {
  __shared__ float tile[32][33];
  const int g  = blockIdx.z;
  const int r0 = blockIdx.x * 32;
  const int c0 = blockIdx.y * 32;
  const int lx = threadIdx.x & 31;
  const int ly = threadIdx.x >> 5;      // 0..7
  const float* s = src + (size_t)g * R * C;
  __bf16*      d = dst + (size_t)g * R * C;
#pragma unroll
  for (int ss = 0; ss < 4; ++ss)
    tile[ly + 8 * ss][lx] = s[(size_t)(r0 + ly + 8 * ss) * C + c0 + lx];
  __syncthreads();
#pragma unroll
  for (int ss = 0; ss < 4; ++ss)
    d[(size_t)(c0 + ly + 8 * ss) * R + r0 + lx] = (__bf16)tile[lx][ly + 8 * ss];
}

__global__ void k_zero(unsigned int* __restrict__ p, size_t n) {
  size_t i = (size_t)blockIdx.x * blockDim.x + threadIdx.x;
  const size_t stride = (size_t)gridDim.x * blockDim.x;
  for (; i < n; i += stride) p[i] = 0u;
}

// --------------------------------------------------------------------------

extern "C" void kernel_launch(void* const* d_in, const int* in_sizes, int n_in,
                              void* d_out, int out_size, void* d_ws, size_t ws_size,
                              hipStream_t stream) {
  (void)in_sizes; (void)n_in; (void)out_size; (void)ws_size;
  const float* v   = (const float*)d_in[0];
  const float* adj = (const float*)d_in[1];
  const float* q   = (const float*)d_in[2];
  const float* Wv  = (const float*)d_in[3];
  const float* bv  = (const float*)d_in[4];
  const float* E   = (const float*)d_in[5];
  const float* mu  = (const float*)d_in[6];
  const float* ls  = (const float*)d_in[7];
  const float* Wk  = (const float*)d_in[8];
  const float* bk  = (const float*)d_in[9];
  float* spa = (float*)d_out;   // spa_v state lives in d_out (f32 [B,N,1024])

  char* p = (char*)d_ws;
  auto carve = [&](size_t bytes) -> void* {
    void* r = (void*)p;
    p += (bytes + 255) & ~(size_t)255;
    return r;
  };
  float*  wg    = (float*)carve((size_t)LDIM * KG * 4);
  float*  maskb = (float*)carve((size_t)BN * 4);
  float*  sg    = (float*)carve((size_t)BN * KG * 4);
  float*  cg    = (float*)carve((size_t)BN * KG * 4);
  float*  qw    = (float*)carve((size_t)B_ * KG * ODIM * 4);
  float*  hq    = (float*)carve((size_t)BN * ODIM * 4);
  __bf16* vbf   = (__bf16*)carve((size_t)BN * VDIM * 2);
  __bf16* wvT   = (__bf16*)carve((size_t)VDIM * ODIM * 2);          // [1024][2048]
  __bf16* wkT   = (__bf16*)carve((size_t)KG * INDIM * ODIM * 2);    // [8][1024][2048]
  __bf16* qbf   = (__bf16*)carve((size_t)B_ * QDIM * 2);
  __bf16* spaT  = (__bf16*)carve((size_t)B_ * ODIM * 64 * 2);       // [128][1024][64]
  __bf16* akn   = (__bf16*)carve((size_t)B_ * KG * 48 * 64 * 2);
  __bf16* ys    = (__bf16*)carve((size_t)B_ * KG * N_ * ODIM * 2);

  // ---- one-time prep ----
  k_w<<<1, 128, 0, stream>>>(E, mu, ls, wg);
  k_cvt<<<2048, 256, 0, stream>>>(v, vbf, (size_t)BN * VDIM);
  k_cvt<<<128,  256, 0, stream>>>(q, qbf, (size_t)B_ * QDIM);
  k_cvtT<<<dim3(VDIM / 32, ODIM / 32, 1),  256, 0, stream>>>(Wv, wvT, VDIM, ODIM);
  k_cvtT<<<dim3(INDIM / 32, ODIM / 32, KG), 256, 0, stream>>>(Wk, wkT, INDIM, ODIM);
  k_zero<<<1024, 256, 0, stream>>>((unsigned int*)akn, (size_t)B_ * KG * 48 * 64 / 2);

  k_gemm_vtrans<<<dim3(BN / 32, 4), 256, 0, stream>>>(vbf, wvT, bv, spa);
  k_gemm_qw<<<dim3(B_ / 32, 4, KG), 256, 0, stream>>>(qbf, wkT, qw);
  k_akn<<<BN, N_ * KG, 0, stream>>>(adj, wg, akn, sg);

  // ---- two propagation steps ----
  for (int step = 0; step < 2; ++step) {
    k_mask<<<BN, 256, 0, stream>>>(spa, maskb);
    k_c<<<B_, N_ * KG, 0, stream>>>(akn, maskb, cg);
    k_hq<<<BN, 256, 0, stream>>>(cg, sg, qw, bk, hq);
    k_spaT<<<dim3(B_, 16), 256, 0, stream>>>(spa, spaT);
    k_gemm_agg<<<dim3(3, 4, B_ * KG), 256, 0, stream>>>(akn, spaT, ys);
    k_gemm_proj<<<dim3(BN / 32, 4), 256, 0, stream>>>(ys, wkT, hq, spa);
  }
}
